// HMM_13383118095070
// MI455X (gfx1250) — matrix operand order repaired and checked by reading, not compiled, not run on hardware
//
#include <hip/hip_runtime.h>

typedef __attribute__((ext_vector_type(16))) __bf16 v16bf;
typedef __attribute__((ext_vector_type(8)))  __bf16 v8bf;
typedef __attribute__((ext_vector_type(8)))  float  v8f;

#define NS    512
#define NOBS  1024
#define BATCH 64
#define TMAX  512
#define AST   520   // alpha LDS row stride (f32), 520 dwords = 8 mod 64 banks
#define WST   520   // W     LDS row stride (bf16)
#define NEG_INF (-3.402823466e+38f)

// ---------- pre-pass: column logsumexp of A (axis=0 softmax) ----------
__global__ void hmm_col_lse(const float* __restrict__ A, float* __restrict__ outc) {
    __shared__ float red[256];
    const int k = blockIdx.x, tid = threadIdx.x;
    float m = NEG_INF;
    for (int i = tid; i < NS; i += 256) m = fmaxf(m, A[i * NS + k]);
    red[tid] = m; __syncthreads();
    for (int s = 128; s; s >>= 1) { if (tid < s) red[tid] = fmaxf(red[tid], red[tid + s]); __syncthreads(); }
    m = red[0]; __syncthreads();
    float sum = 0.f;
    for (int i = tid; i < NS; i += 256) sum += expf(A[i * NS + k] - m);
    red[tid] = sum; __syncthreads();
    for (int s = 128; s; s >>= 1) { if (tid < s) red[tid] += red[tid + s]; __syncthreads(); }
    if (tid == 0) outc[k] = m + logf(red[0]);
}

// ---------- pre-pass: row logsumexp of E (axis=1 softmax) ----------
__global__ void hmm_row_lse(const float* __restrict__ E, float* __restrict__ outr) {
    __shared__ float red[256];
    const int i = blockIdx.x, tid = threadIdx.x;
    float m = NEG_INF;
    for (int c = tid; c < NOBS; c += 256) m = fmaxf(m, E[i * NOBS + c]);
    red[tid] = m; __syncthreads();
    for (int s = 128; s; s >>= 1) { if (tid < s) red[tid] = fmaxf(red[tid], red[tid + s]); __syncthreads(); }
    m = red[0]; __syncthreads();
    float sum = 0.f;
    for (int c = tid; c < NOBS; c += 256) sum += expf(E[i * NOBS + c] - m);
    red[tid] = sum; __syncthreads();
    for (int s = 128; s; s >>= 1) { if (tid < s) red[tid] += red[tid + s]; __syncthreads(); }
    if (tid == 0) outr[i] = m + logf(red[0]);
}

// ---------- pre-pass: logsumexp of pi ----------
__global__ void hmm_pi_lse(const float* __restrict__ pi, float* __restrict__ outp) {
    __shared__ float red[512];
    const int tid = threadIdx.x;
    red[tid] = pi[tid]; __syncthreads();
    for (int s = 256; s; s >>= 1) { if (tid < s) red[tid] = fmaxf(red[tid], red[tid + s]); __syncthreads(); }
    const float m = red[0]; __syncthreads();
    red[tid] = expf(pi[tid] - m); __syncthreads();
    for (int s = 256; s; s >>= 1) { if (tid < s) red[tid] += red[tid + s]; __syncthreads(); }
    if (tid == 0) outp[0] = m + logf(red[0]);
}

// ---------- pre-pass: pack P = exp(log_softmax(A,axis=0)) into WMMA-B swizzle ----------
// B operand (32x16 bf16, K x N): lane n = N = lane&15; K = (lane&16) + j, j = 0..15
// contiguous per lane. Tile index = itile*16 + ktile, 512 bf16/tile.
__global__ void hmm_packP(const float* __restrict__ A, const float* __restrict__ colLSE,
                          __bf16* __restrict__ Pswz) {
    const int f = blockIdx.x * blockDim.x + threadIdx.x;   // 0 .. 512*512-1
    const int j     = f & 15;
    const int lane  = (f >> 4) & 31;
    const int tile  = f >> 9;
    const int ktile = tile & 15;
    const int itile = tile >> 4;
    const int i = itile * 16 + (lane & 15);
    const int k = ktile * 32 + (lane & 16) + j;
    const float la = A[i * NS + k] - colLSE[k];
    Pswz[f] = (__bf16)expf(la);
}

// ---------- main: sequential forward scan; 1 WG = 16 batch rows, 32 waves = 32 i-tiles ----------
__global__ __launch_bounds__(1024, 1)
void hmm_forward(const float* __restrict__ E, const float* __restrict__ pi,
                 const int* __restrict__ x, const int* __restrict__ Tlen,
                 const float* __restrict__ rowLSE, const float* __restrict__ piLSE,
                 const __bf16* __restrict__ Pswz, float* __restrict__ out) {
    __shared__ float  alpha[16 * AST];   // (b, i) log-alpha, f32
    __shared__ __bf16 Wl[16 * WST];      // (b, k) exp(alpha - m), bf16
    __shared__ float  mrow[16];
    __shared__ float  red32[32];
    __shared__ int    obsN[16];          // x[b][t+1]

    const int wg   = blockIdx.x;
    const int tid  = threadIdx.x;
    const int wave = tid >> 5;
    const int lane = tid & 31;

    const float piL = piLSE[0];

    // ---- init: alpha0[b,i] = log_E[i, x[b,0]] + log_pi[i] ----
    for (int f = tid; f < 16 * NS; f += 1024) {
        const int b = f >> 9, i = f & (NS - 1);
        const int bg = wg * 16 + b;
        const int obs0 = x[bg * TMAX];
        alpha[b * AST + i] = (E[i * NOBS + obs0] - rowLSE[i]) + (pi[i] - piL);
    }

    // per-wave GEMM constants
    const int itile = wave;
    const int lb    = lane & 15;           // M row (A) / N col (B,C)
    const int hiOff = (lane & 16) >> 1;    // 0 or 8
    const int icol  = itile * 16 + lb;     // output state index i for this lane
    const float rlse_i = rowLSE[icol];

    for (int t = 0; t < TMAX; ++t) {
        __syncthreads();   // alpha (init or previous step's GEMM) is ready

        // ---- Phase A: m[b] = max_k alpha[b,k] (waves 0..15); prefetch obs[t+1] (wave 16) ----
        if (wave < 16) {
            const int b = wave;
            float m = NEG_INF;
            #pragma unroll
            for (int j = 0; j < 16; ++j)
                m = fmaxf(m, alpha[b * AST + lane + 32 * j]);
            #pragma unroll
            for (int off = 16; off >= 1; off >>= 1)
                m = fmaxf(m, __shfl_xor(m, off, 32));
            if (lane == 0) mrow[b] = m;
        } else if (wave == 16 && lane < 16) {
            obsN[lane] = (t + 1 < TMAX) ? x[(wg * 16 + lane) * TMAX + (t + 1)] : 0;
        }
        __syncthreads();

        // ---- Phase B: W[b,k] = exp(alpha - m) -> bf16 LDS; row sums for logsumexp ----
        {
            const int b  = wave >> 1;
            const int kb = (wave & 1) * 256 + lane;
            const float m = mrow[b];
            float s = 0.f;
            #pragma unroll
            for (int j = 0; j < 8; ++j) {
                const int k = kb + 32 * j;
                const float e = __expf(alpha[b * AST + k] - m);
                Wl[b * WST + k] = (__bf16)e;
                s += e;
            }
            #pragma unroll
            for (int off = 16; off >= 1; off >>= 1)
                s += __shfl_xor(s, off, 32);
            if (lane == 0) red32[wave] = s;
        }
        __syncthreads();

        // ---- emit output for batches finishing at this t ----
        if (tid < 16) {
            const float stot = red32[2 * tid] + red32[2 * tid + 1];
            if (t == Tlen[wg * 16 + tid] - 1)
                out[wg * 16 + tid] = mrow[tid] + __logf(stot);
        }

        // ---- Phase C: GEMM S = W x P^T via WMMA bf16, then alpha_{t+1} ----
        if (t < TMAX - 1) {
            v8f c = {};
            #pragma unroll
            for (int kt = 0; kt < 16; ++kt) {
                const int kbase = kt * 32;
                // A operand 16x32 bf16: lane<16 -> K {0..15}; lane>=16 -> K {16..31}
                const v8bf a0 = *(const v8bf*)&Wl[lb * WST + kbase + 2 * hiOff];
                const v8bf a1 = *(const v8bf*)&Wl[lb * WST + kbase + 2 * hiOff + 8];
                const v16bf amat = __builtin_shufflevector(a0, a1,
                    0, 1, 2, 3, 4, 5, 6, 7, 8, 9, 10, 11, 12, 13, 14, 15);
                // B operand: pre-swizzled, 32 contiguous bytes per lane
                const v16bf bmat = *(const v16bf*)(Pswz +
                    ((size_t)(itile * 16 + kt)) * 512 + lane * 16);
                c = __builtin_amdgcn_wmma_f32_16x16x32_bf16(
                        false, amat, false, bmat, (short)0, c, false, false);
            }
            // C/D layout: VGPR r -> (b = r + hiOff, i = icol)
            #pragma unroll
            for (int r = 0; r < 8; ++r) {
                const int b = r + hiOff;
                const float emis = E[icol * NOBS + obsN[b]] - rlse_i;
                alpha[b * AST + icol] = emis + mrow[b] + __logf(c[r]);
            }
        }
    }
}

extern "C" void kernel_launch(void* const* d_in, const int* in_sizes, int n_in,
                              void* d_out, int out_size, void* d_ws, size_t ws_size,
                              hipStream_t stream) {
    (void)in_sizes; (void)n_in; (void)out_size; (void)ws_size;
    const float* pi = (const float*)d_in[0];
    const float* A  = (const float*)d_in[1];
    const float* E  = (const float*)d_in[2];
    const int*   x  = (const int*)d_in[3];
    const int*   Tl = (const int*)d_in[4];
    float* out = (float*)d_out;

    // workspace: [Pswz bf16 512KB][colLSE 2KB][rowLSE 2KB][piLSE 4B]
    __bf16* Pswz = (__bf16*)d_ws;
    float* colL = (float*)((char*)d_ws + (size_t)NS * NS * 2);
    float* rowL = colL + NS;
    float* piL  = rowL + NS;

    hmm_col_lse<<<NS, 256, 0, stream>>>(A, colL);
    hmm_row_lse<<<NS, 256, 0, stream>>>(E, rowL);
    hmm_pi_lse<<<1, 512, 0, stream>>>(pi, piL);
    hmm_packP<<<(NS * NS) / 256, 256, 0, stream>>>(A, colL, Pswz);
    hmm_forward<<<BATCH / 16, 1024, 0, stream>>>(E, pi, x, Tl, rowL, piL, Pswz, out);
}
